// Beamformer_Network_8873402433740
// MI455X (gfx1250) — matrix operand order repaired and checked by reading, not compiled, not run on hardware
//
#include <hip/hip_runtime.h>
#include <math.h>

// ---------------------------------------------------------------------------
// CDNA5 (gfx1250) wave32 WMMA implementation of the beamformer GNN.
// All dense layers run through one tiled bf16 WMMA GEMM kernel:
//   C = epilogue( A(bf16)[M,K] @ W(bf16)[K,N] + bias(f32)[N] )
// A tiles are DMA'd into LDS with global_load_async_to_lds_b128 (ASYNCcnt),
// double-buffered and software-pipelined under the WMMAs. W tiles are
// register-staged one iteration ahead and scattered transposed into LDS.
// ---------------------------------------------------------------------------

typedef unsigned short u16t;
typedef __attribute__((ext_vector_type(16))) __bf16 v16bf;
typedef __attribute__((ext_vector_type(8)))  float  v8f;

#define Bsz 4096
#define Ssz 128
#define Fsz 128
#define BB2 64
#define RF2 512

__device__ __forceinline__ u16t f2bf(float f) {
  unsigned u = __float_as_uint(f);
  u += 0x7FFFu + ((u >> 16) & 1u);          // round-to-nearest-even
  return (u16t)(u >> 16);
}
__device__ __forceinline__ float bf2f(u16t h) {
  return __uint_as_float(((unsigned)h) << 16);
}
__device__ __forceinline__ float mishf(float x) {
  float sp = (x > 20.0f) ? x : log1pf(__expf(x));
  return x * tanhf(sp);
}

// fp32 -> bf16 conversion (weights + feedback), run once per call
__global__ void cvt_f32_bf16(const float* __restrict__ in,
                             u16t* __restrict__ out, int n) {
  int i = blockIdx.x * 256 + threadIdx.x;
  if (i < n) out[i] = f2bf(in[i]);
}

// BB_mean[b, j] = mean over s of BB[b, s, j]   (bf16 in/out, f32 accumulate)
__global__ void mean_over_s(const u16t* __restrict__ BB,
                            u16t* __restrict__ outm) {
  int idx = blockIdx.x * 256 + threadIdx.x;   // b*64 + j
  int b = idx >> 6, j = idx & 63;
  const u16t* p = BB + (size_t)b * (Ssz * BB2) + j;
  float s = 0.f;
#pragma unroll 4
  for (int si = 0; si < Ssz; ++si) s += bf2f(p[(size_t)si * BB2]);
  outm[idx] = f2bf(s * (1.0f / 128.0f));
}

// ---------------------------------------------------------------------------
// Tiled WMMA GEMM.  Block tile 128x64, BK=64, 256 threads = 8 wave32 waves.
// Wave (wm, wn) owns a 32x32 tile = 2x2 frags of v_wmma_f32_16x16x32_bf16.
// Output addressing:
//   off = z*sCz + (row/rdiv)*ldcO + (row%rdiv)*ldcI + colOff + col
// ---------------------------------------------------------------------------
template <bool OUT_F32, bool MISH>
__global__ __launch_bounds__(256) void wmma_gemm(
    const u16t* __restrict__ A, int lda, long sAz,
    const u16t* __restrict__ W, int ldw, long sWz,
    const float* __restrict__ bias, int sBz,
    const u16t* __restrict__ addv, int addLd, int addDiv,
    void* __restrict__ outp, long sCz,
    int ldcO, int ldcI, int rdiv, int colOff,
    int K, int N) {
  constexpr int BMt = 128, BNt = 64, BKt = 64, LDSP = BKt + 8;  // 144B rows
  __shared__ u16t sA[2][BMt * LDSP];
  __shared__ u16t sB[BNt * LDSP];

  const int z = blockIdx.z;
  A += (long)z * sAz;
  W += (long)z * sWz;
  bias += (long)z * sBz;
  const long outZ = (long)z * sCz;

  const int tid = threadIdx.x, lane = tid & 31, wave = tid >> 5;
  const int waveM = wave & 3, waveN = wave >> 2;   // 4x2 wave grid
  const int rowBase = blockIdx.x * BMt, colBase = blockIdx.y * BNt;
  const int hi = lane >> 4, lr = lane & 15;

  // -- A tile: 4 async 16B DMA chunks per thread, global layout == LDS layout
  auto issueA = [&](int k0, int buf) {
#pragma unroll
    for (int i = 0; i < 4; ++i) {
      int c = tid + i * 256;        // 0..1023
      int r = c >> 3;               // row 0..127
      int kc = (c & 7) * 8;         // k-chunk of 8 bf16
      const u16t* gp = A + (size_t)(rowBase + r) * lda + k0 + kc;
      unsigned lo = (unsigned)(size_t)(&sA[buf][r * LDSP + kc]);
      asm volatile("global_load_async_to_lds_b128 %0, %1, off"
                   :: "v"(lo), "v"(gp) : "memory");
    }
  };

  // -- W tile: 2 x b128 register loads per thread (issued a tile ahead),
  //    scattered transposed ([n][k]) into sB after the barrier.
  uint4 wreg[2];
  auto loadW = [&](int k0) {
#pragma unroll
    for (int i = 0; i < 2; ++i) {
      int e = tid + i * 256;        // 0..511
      int k = e >> 3;               // 0..63
      int nc = (e & 7) * 8;         // n-chunk of 8
      if (colBase + nc + 8 <= N) {
        wreg[i] = *(const uint4*)&W[(size_t)(k0 + k) * ldw + colBase + nc];
      } else {                       // ragged tile (N=16 heads): zero-pad
        u16t tmp[8];
#pragma unroll
        for (int j = 0; j < 8; ++j)
          tmp[j] = (colBase + nc + j < N)
                       ? W[(size_t)(k0 + k) * ldw + colBase + nc + j]
                       : (u16t)0;
        wreg[i] = *(uint4*)tmp;
      }
    }
  };
  auto storeW = [&]() {
#pragma unroll
    for (int i = 0; i < 2; ++i) {
      int e = tid + i * 256;
      int k = e >> 3;
      int nc = (e & 7) * 8;
      const u16t* src = (const u16t*)&wreg[i];
#pragma unroll
      for (int j = 0; j < 8; ++j) sB[(nc + j) * LDSP + k] = src[j];
    }
  };

  v8f acc[2][2];
#pragma unroll
  for (int i = 0; i < 2; ++i)
#pragma unroll
    for (int j = 0; j < 2; ++j)
#pragma unroll
      for (int e = 0; e < 8; ++e) acc[i][j][e] = 0.0f;

  // ---- software pipeline: prologue ----
  issueA(0, 0);
  loadW(0);

  int buf = 0;
  for (int k0 = 0; k0 < K; k0 += BKt) {
    const bool hasNext = (k0 + BKt) < K;
    __syncthreads();                 // all waves done reading sA[buf^1], sB
    if (hasNext) issueA(k0 + BKt, buf ^ 1);   // DMA next A tile under compute
    storeW();                        // commit current W tile to LDS
    if (hasNext)                     // retire current tile's 4 async ops only
      asm volatile("s_wait_asynccnt 0x4" ::: "memory");
    else
      asm volatile("s_wait_asynccnt 0x0" ::: "memory");
    __syncthreads();                 // sA[buf] + sB visible to all waves
    if (hasNext) loadW(k0 + BKt);    // next W tile -> regs (latency hidden)

#pragma unroll
    for (int ks = 0; ks < BKt; ks += 32) {
      v16bf af[2], bfr[2];
#pragma unroll
      for (int fm = 0; fm < 2; ++fm) {
        // A frag 16x32: lane lr row, hi selects K-groups {0..7,16..23}/{8..15,24..31}
        const u16t* base = &sA[buf][(waveM * 32 + fm * 16 + lr) * LDSP + ks];
        uint4* dst = reinterpret_cast<uint4*>(&af[fm]);
        dst[0] = *(const uint4*)(base + hi * 8);
        dst[1] = *(const uint4*)(base + 16 + hi * 8);
      }
#pragma unroll
      for (int fn = 0; fn < 2; ++fn) {
        // B frag 32x16: lane lr column, hi selects K-groups {0..7,16..23}/{8..15,24..31}
        const u16t* base =
            &sB[(waveN * 32 + fn * 16 + lr) * LDSP + ks];
        uint4* dst = reinterpret_cast<uint4*>(&bfr[fn]);
        dst[0] = *(const uint4*)(base + hi * 8);
        dst[1] = *(const uint4*)(base + 16 + hi * 8);
      }
#pragma unroll
      for (int fm = 0; fm < 2; ++fm)
#pragma unroll
        for (int fn = 0; fn < 2; ++fn)
          acc[fm][fn] = __builtin_amdgcn_wmma_f32_16x16x32_bf16(
              false, af[fm], false, bfr[fn], (short)0, acc[fm][fn], false,
              false);
    }
    buf ^= 1;
  }

  // ---- epilogue: bias + Mish + optional broadcast-add, strided store ----
  float* of = (float*)outp;
  u16t* ob = (u16t*)outp;
#pragma unroll
  for (int fm = 0; fm < 2; ++fm)
#pragma unroll
    for (int fn = 0; fn < 2; ++fn) {
      int gcol = colBase + waveN * 32 + fn * 16 + lr;
      if (gcol >= N) continue;
      float bv = bias[gcol];
#pragma unroll
      for (int vr = 0; vr < 8; ++vr) {
        int grow = rowBase + waveM * 32 + fm * 16 + vr + hi * 8;
        float x = acc[fm][fn][vr] + bv;
        if (MISH) x = mishf(x);
        if (addv) x += bf2f(addv[(size_t)(grow / addDiv) * addLd + gcol]);
        size_t off = (size_t)outZ + (size_t)(grow / rdiv) * ldcO +
                     (size_t)(grow % rdiv) * ldcI + colOff + gcol;
        if (OUT_F32) of[off] = x;
        else         ob[off] = f2bf(x);
      }
    }
}

// ---------------------------------------------------------------------------
// Host-side dispatch
// ---------------------------------------------------------------------------
static inline void gemm(hipStream_t st, const u16t* A, int lda, long sAz,
                        const u16t* W, int ldw, long sWz, const float* bias,
                        int sBz, const u16t* addv, int addLd, int addDiv,
                        void* outp, long sCz, int ldcO, int ldcI, int rdiv,
                        int colOff, int M, int K, int N, int Z, bool f32out) {
  dim3 g(M / 128, (N + 63) / 64, Z), b(256, 1, 1);
  if (f32out)
    wmma_gemm<true, false><<<g, b, 0, st>>>(A, lda, sAz, W, ldw, sWz, bias,
                                            sBz, addv, addLd, addDiv, outp,
                                            sCz, ldcO, ldcI, rdiv, colOff, K, N);
  else
    wmma_gemm<false, true><<<g, b, 0, st>>>(A, lda, sAz, W, ldw, sWz, bias,
                                            sBz, addv, addLd, addDiv, outp,
                                            sCz, ldcO, ldcI, rdiv, colOff, K, N);
}

extern "C" void kernel_launch(void* const* d_in, const int* in_sizes, int n_in,
                              void* d_out, int out_size, void* d_ws,
                              size_t ws_size, hipStream_t stream) {
  (void)in_sizes; (void)n_in; (void)out_size; (void)ws_size;
  u16t* ws = (u16t*)d_ws;
  size_t cur = 0;
  auto alloc = [&](size_t n) {
    u16t* p = ws + cur;
    cur += (n + 127) & ~(size_t)127;
    return p;
  };

  // ---- fp32 -> bf16: feedback + all 14 weight tensors ----
  const int    ci[15] = {0, 1, 3, 5, 7, 9, 11, 13, 15, 17, 19, 21, 23, 25, 27};
  const size_t cn[15] = {524288, 1048576, 524288, 65536, 262144, 786432, 98304,
                         98304, 12288, 524288, 262144, 8192, 2048, 8192, 2048};
  u16t* cb[29] = {};
  for (int i = 0; i < 15; ++i) {
    cb[ci[i]] = alloc(cn[i]);
    int n = (int)cn[i];
    cvt_f32_bf16<<<dim3((n + 255) / 256), dim3(256), 0, stream>>>(
        (const float*)d_in[ci[i]], cb[ci[i]], n);
  }

  // ---- workspace buffers (bf16) ----
  u16t* BBa  = alloc((size_t)Bsz * Ssz * BB2);
  u16t* BBb  = alloc((size_t)Bsz * Ssz * BB2);
  u16t* RFa  = alloc((size_t)Bsz * RF2);
  u16t* RFb  = alloc((size_t)Bsz * RF2);
  u16t* RFt  = alloc((size_t)Bsz * RF2);
  u16t* RFBB = alloc((size_t)Bsz * BB2);
  u16t* BBm  = alloc((size_t)Bsz * BB2);
  u16t* Ahid = alloc((size_t)Bsz * 1024);
  u16t* Dhid = alloc((size_t)Bsz * Ssz * 128);

  const int SB = Ssz * BB2;  // 8192, BB row stride

  // ---- init BB branch: per-subcarrier (z = s) batched GEMMs ----
  gemm(stream, cb[0], Fsz, 0, cb[1], BB2, (long)Fsz * BB2,
       (const float*)d_in[2], BB2, nullptr, 1, 1, BBa, BB2, SB, 0, 1, 0, Bsz,
       Fsz, BB2, Ssz, false);
  gemm(stream, BBa, SB, BB2, cb[3], BB2, (long)BB2 * BB2,
       (const float*)d_in[4], BB2, nullptr, 1, 1, BBb, BB2, SB, 0, 1, 0, Bsz,
       BB2, BB2, Ssz, false);

  // ---- init RF branch ----
  gemm(stream, cb[0], Fsz, 0, cb[5], RF2, 0, (const float*)d_in[6], 0, nullptr,
       1, 1, RFt, 0, RF2, 0, 1, 0, Bsz, Fsz, RF2, 1, false);
  gemm(stream, RFt, RF2, 0, cb[7], RF2, 0, (const float*)d_in[8], 0, nullptr,
       1, 1, RFa, 0, RF2, 0, 1, 0, Bsz, RF2, RF2, 1, false);

  // ---- 3 message-passing layers ----
  u16t *BBcur = BBb, *BBnxt = BBa, *RFcur = RFa, *RFnxt = RFb;
  for (int l = 0; l < 3; ++l) {
    mean_over_s<<<dim3(Bsz * BB2 / 256), dim3(256), 0, stream>>>(BBcur, BBm);
    // RF_to_RF = mish(RF @ rfrf)
    gemm(stream, RFcur, RF2, 0, cb[9] + (size_t)l * RF2 * RF2, RF2, 0,
         (const float*)d_in[10] + l * RF2, 0, nullptr, 1, 1, RFt, 0, RF2, 0, 1,
         0, Bsz, RF2, RF2, 1, false);
    // RF' = mish(BBmean @ bbrf) + RF_to_RF
    gemm(stream, BBm, BB2, 0, cb[13] + (size_t)l * BB2 * RF2, RF2, 0,
         (const float*)d_in[14] + l * RF2, 0, RFt, RF2, 1, RFnxt, 0, RF2, 0, 1,
         0, Bsz, BB2, RF2, 1, false);
    // RF_to_BB = mish(RF @ rfbb)
    gemm(stream, RFcur, RF2, 0, cb[11] + (size_t)l * RF2 * BB2, BB2, 0,
         (const float*)d_in[12] + l * BB2, 0, nullptr, 1, 1, RFBB, 0, BB2, 0,
         1, 0, Bsz, RF2, BB2, 1, false);
    // BB' = mish(BB @ bbbb) + RF_to_BB[b] broadcast over s  (flattened B*S)
    gemm(stream, BBcur, BB2, 0, cb[15] + (size_t)l * BB2 * BB2, BB2, 0,
         (const float*)d_in[16] + l * BB2, 0, RFBB, BB2, Ssz, BBnxt, 0, BB2, 0,
         1, 0, Bsz * Ssz, BB2, BB2, 1, false);
    u16t* t = BBcur; BBcur = BBnxt; BBnxt = t;
    t = RFcur; RFcur = RFnxt; RFnxt = t;
  }

  // ---- heads: write f32 directly into d_out [B, 256 | 2048 | 2048] ----
  const int LDO = 4352;
  // analog: RF -> 1024 (mish) -> 256
  gemm(stream, RFcur, RF2, 0, cb[17], 1024, 0, (const float*)d_in[18], 0,
       nullptr, 1, 1, Ahid, 0, 1024, 0, 1, 0, Bsz, RF2, 1024, 1, false);
  gemm(stream, Ahid, 1024, 0, cb[19], 256, 0, (const float*)d_in[20], 0,
       nullptr, 1, 1, d_out, 0, LDO, 0, 1, 0, Bsz, 1024, 256, 1, true);
  // digital real: BB -> 128 (mish) -> 16, scatter to cols [256, 2304)
  gemm(stream, BBcur, BB2, 0, cb[21], 128, 0, (const float*)d_in[22], 0,
       nullptr, 1, 1, Dhid, 0, 128, 0, 1, 0, Bsz * Ssz, BB2, 128, 1, false);
  gemm(stream, Dhid, 128, 0, cb[23], 16, 0, (const float*)d_in[24], 0, nullptr,
       1, 1, d_out, 0, LDO, 16, Ssz, 256, Bsz * Ssz, 128, 16, 1, true);
  // digital imag: scatter to cols [2304, 4352)
  gemm(stream, BBcur, BB2, 0, cb[25], 128, 0, (const float*)d_in[26], 0,
       nullptr, 1, 1, Dhid, 0, 128, 0, 1, 0, Bsz * Ssz, BB2, 128, 1, false);
  gemm(stream, Dhid, 128, 0, cb[27], 16, 0, (const float*)d_in[28], 0, nullptr,
       1, 1, d_out, 0, LDO, 16, Ssz, 2304, Bsz * Ssz, 128, 16, 1, true);
}
